// VAEHyperNetFusion_29437705847408
// MI455X (gfx1250) — compile-verified, matrix operand order checked by reading, not run on hardware
//
#include <hip/hip_runtime.h>

typedef __bf16 bf16_t;
typedef bf16_t v16bf __attribute__((ext_vector_type(16)));
typedef float  v8f   __attribute__((ext_vector_type(8)));
typedef unsigned short u16;

#define NROWS      32768
#define IN_DIM     512
#define NSUB       20
#define NCLS       100
#define NCLS_PAD   112
#define SUBH       32
#define TOTAL      19716   // 16384 + 32 + 3200 + 100

// workspace layout (bytes), all 32B-aligned
#define OFF_XBF    0ull                 // 32768*512*2      = 33,554,432
#define OFF_PART   33554432ull          // 256*512*4        =    524,288
#define OFF_H2     34078720ull          // 256*4
#define OFF_WSM    34079744ull          // 20*4 (pad 128)
#define OFF_W1BF   34079872ull          // 20*32*512*2      =    655,360
#define OFF_B1F    34735232ull          // 20*32*4          =      2,560
#define OFF_W2BF   34737792ull          // 20*112*32*2      =    143,360
#define OFF_B2F    34881152ull          // 20*100*4

__device__ __forceinline__ u16 f2bf(float f) {
    unsigned int u = __float_as_uint(f);
    unsigned int r = u + 0x7FFFu + ((u >> 16) & 1u);   // round-to-nearest-even
    return (u16)(r >> 16);
}

// ---- kA: column partial sums (for mean) + x -> bf16 conversion ----
__global__ __launch_bounds__(512)
void kA(const float* __restrict__ x, u16* __restrict__ xbf, float* __restrict__ part) {
    int col = threadIdx.x;                 // 512 cols
    int rb  = blockIdx.x * 128;            // 256 blocks * 128 rows
    float s = 0.f;
    for (int r = 0; r < 128; ++r) {
        float v = x[(size_t)(rb + r) * IN_DIM + col];
        s += v;
        xbf[(size_t)(rb + r) * IN_DIM + col] = f2bf(v);
    }
    part[blockIdx.x * IN_DIM + col] = s;
}

// ---- kB: hypernet MLP (512 -> 128 -> 256) + layernorms + softmax(ens_w) ----
__global__ __launch_bounds__(512)
void kB(const float* __restrict__ part,
        const float* __restrict__ Wh1, const float* __restrict__ bh1,
        const float* __restrict__ g1,  const float* __restrict__ be1,
        const float* __restrict__ Wh2, const float* __restrict__ bh2,
        const float* __restrict__ g2,  const float* __restrict__ be2,
        const float* __restrict__ ensw, float* __restrict__ h2out,
        float* __restrict__ wsm) {
    __shared__ float ctx[512];
    __shared__ float tmp[256];
    __shared__ float h1s[128];
    __shared__ float red[2];
    int tid = threadIdx.x;

    float s = 0.f;
    for (int p = 0; p < 256; ++p) s += part[p * IN_DIM + tid];
    ctx[tid] = s * (1.0f / (float)NROWS);
    __syncthreads();

    if (tid < 128) {
        float a = bh1[tid];
        for (int c = 0; c < 512; ++c) a += ctx[c] * Wh1[tid * 512 + c];
        tmp[tid] = a;
    }
    __syncthreads();
    if (tid == 0) {
        float mu = 0.f; for (int i = 0; i < 128; ++i) mu += tmp[i];
        mu *= (1.f/128.f);
        float va = 0.f; for (int i = 0; i < 128; ++i) { float d = tmp[i]-mu; va += d*d; }
        va *= (1.f/128.f);
        red[0] = mu; red[1] = rsqrtf(va + 1e-5f);
    }
    __syncthreads();
    if (tid < 128) h1s[tid] = fmaxf(0.f, (tmp[tid]-red[0])*red[1]*g1[tid] + be1[tid]);
    __syncthreads();

    if (tid < 256) {
        float a = bh2[tid];
        for (int c = 0; c < 128; ++c) a += h1s[c] * Wh2[tid * 128 + c];
        tmp[tid] = a;
    }
    __syncthreads();
    if (tid == 0) {
        float mu = 0.f; for (int i = 0; i < 256; ++i) mu += tmp[i];
        mu *= (1.f/256.f);
        float va = 0.f; for (int i = 0; i < 256; ++i) { float d = tmp[i]-mu; va += d*d; }
        va *= (1.f/256.f);
        red[0] = mu; red[1] = rsqrtf(va + 1e-5f);
    }
    __syncthreads();
    if (tid < 256) h2out[tid] = fmaxf(0.f, (tmp[tid]-red[0])*red[1]*g2[tid] + be2[tid]);

    if (tid == 0) {                         // softmax over 20 ensemble weights
        float mx = ensw[0];
        for (int i = 1; i < NSUB; ++i) mx = fmaxf(mx, ensw[i]);
        float e[NSUB]; float sm = 0.f;
        for (int i = 0; i < NSUB; ++i) { e[i] = __expf(ensw[i]-mx); sm += e[i]; }
        float inv = 1.f / sm;
        for (int i = 0; i < NSUB; ++i) wsm[i] = e[i] * inv;
    }
}

// ---- kC: all_params = h2 @ Wh3^T + bh3, scattered into bf16 W1/W2 + f32 b1/b2 ----
__global__ __launch_bounds__(256)
void kC(const float* __restrict__ Wh3, const float* __restrict__ bh3,
        const float* __restrict__ h2, u16* __restrict__ W1bf, float* __restrict__ b1f,
        u16* __restrict__ W2bf, float* __restrict__ b2f) {
    __shared__ float h2s[256];
    int tid = threadIdx.x;
    if (tid < 256) h2s[tid] = h2[tid];
    __syncthreads();
    int wv = tid >> 5, lane = tid & 31;
    long r = (long)blockIdx.x * 8 + wv;          // one wave per output row
    if (r >= (long)NSUB * TOTAL) return;

    const float* row = Wh3 + r * 256 + lane * 8;  // 32B/lane, fully coalesced
    float acc = 0.f;
    #pragma unroll
    for (int i = 0; i < 8; ++i) acc += row[i] * h2s[lane * 8 + i];
    #pragma unroll
    for (int m = 16; m >= 1; m >>= 1) acc += __shfl_xor(acc, m, 32);

    if (lane == 0) {
        float v = acc + bh3[r];
        int s = (int)(r / TOTAL);
        int o = (int)(r % TOTAL);
        if (o < 16384)       W1bf[s * 16384 + o] = f2bf(v);
        else if (o < 16416)  b1f[s * 32 + (o - 16384)] = v;
        else if (o < 19616) {
            int o3 = o - 16416; int c = o3 >> 5; int h = o3 & 31;
            W2bf[(s * NCLS_PAD + c) * 32 + h] = f2bf(v);
        } else               b2f[s * NCLS + (o - 19616)] = v;
    }
}

// ---- kD: fused 20-subnet batched MLP via bf16 WMMA, weighted-sum ensemble ----
__global__ __launch_bounds__(128)
void kD(const u16* __restrict__ xbf, const u16* __restrict__ W1bf,
        const float* __restrict__ b1f, const u16* __restrict__ W2bf,
        const float* __restrict__ b2f, const float* __restrict__ wsm,
        float* __restrict__ out) {
    __shared__ __attribute__((aligned(32))) u16 hhbuf[4][16][32];
    __shared__ float b2w[NCLS_PAD];
    __shared__ float wsms[NSUB];
    int tid = threadIdx.x;
    if (tid < NCLS_PAD) {                    // ensemble-weighted output bias
        float v = 0.f;
        if (tid < NCLS) for (int s = 0; s < NSUB; ++s) v += wsm[s] * b2f[s * NCLS + tid];
        b2w[tid] = v;
    }
    if (tid < NSUB) wsms[tid] = wsm[tid];
    __syncthreads();

    int wv = tid >> 5, lane = tid & 31;
    int lm = lane & 15, lh = lane >> 4;      // lh selects K 0-15 vs 16-31 half
    int mBase = blockIdx.x * 64 + wv * 16;   // 16 sample rows per wave

    const u16* arow = xbf + ((size_t)(mBase + lm) * IN_DIM + lh * 16);

    v8f accC[7];
    #pragma unroll
    for (int i = 0; i < 7; ++i) accC[i] = v8f{};

    for (int s = 0; s < NSUB; ++s) {
        // stage 1: hh(16x32) = x_tile(16x512) @ W1[s]^T, K in 16 chunks of 32
        v8f acch0 = v8f{}, acch1 = v8f{};
        const u16* b1base = W1bf + (size_t)s * 16384 + (size_t)lm * IN_DIM + lh * 16;
        #pragma unroll 4
        for (int kc = 0; kc < 16; ++kc) {
            v16bf av  = *(const v16bf*)(arow  + kc * 32);
            v16bf bv0 = *(const v16bf*)(b1base + kc * 32);
            v16bf bv1 = *(const v16bf*)(b1base + 16 * IN_DIM + kc * 32);
            acch0 = __builtin_amdgcn_wmma_f32_16x16x32_bf16(false, av, false, bv0,
                                                            (short)0, acch0, false, false);
            acch1 = __builtin_amdgcn_wmma_f32_16x16x32_bf16(false, av, false, bv1,
                                                            (short)0, acch1, false, false);
        }
        // bias + relu, fold softmax weight into hh; C-layout -> A-layout via LDS
        float wgt = wsms[s];
        float bb0 = b1f[s * 32 + lm];
        float bb1 = b1f[s * 32 + 16 + lm];
        #pragma unroll
        for (int r = 0; r < 8; ++r) {
            int m = r + 8 * lh;
            hhbuf[wv][m][lm]      = f2bf(fmaxf(0.f, acch0[r] + bb0) * wgt);
            hhbuf[wv][m][16 + lm] = f2bf(fmaxf(0.f, acch1[r] + bb1) * wgt);
        }
        asm volatile("s_wait_dscnt 0" ::: "memory");   // wave-private LDS: DS in-order, just fence
        v16bf a2 = *(const v16bf*)(&hhbuf[wv][lm][lh * 16]);
        asm volatile("" ::: "memory");                 // keep next-iter stores after this load

        // stage 2: logits(16x112) += hh(16x32) @ W2[s]^T (padded), one K=32 WMMA per tile
        const u16* b2base = W2bf + (size_t)s * NCLS_PAD * 32 + (size_t)lm * 32 + lh * 16;
        #pragma unroll
        for (int ct = 0; ct < 7; ++ct) {
            v16bf bv = *(const v16bf*)(b2base + (size_t)ct * 16 * 32);
            accC[ct] = __builtin_amdgcn_wmma_f32_16x16x32_bf16(false, a2, false, bv,
                                                               (short)0, accC[ct], false, false);
        }
    }

    #pragma unroll
    for (int ct = 0; ct < 7; ++ct) {
        #pragma unroll
        for (int r = 0; r < 8; ++r) {
            int m = mBase + r + 8 * lh;
            int c = ct * 16 + lm;
            if (c < NCLS) out[(size_t)m * NCLS + c] = accC[ct][r] + b2w[c];
        }
    }
}

extern "C" void kernel_launch(void* const* d_in, const int* in_sizes, int n_in,
                              void* d_out, int out_size, void* d_ws, size_t ws_size,
                              hipStream_t stream) {
    const float* x    = (const float*)d_in[0];
    const float* Wh1  = (const float*)d_in[1];
    const float* bh1  = (const float*)d_in[2];
    const float* g1   = (const float*)d_in[3];
    const float* be1  = (const float*)d_in[4];
    const float* Wh2  = (const float*)d_in[5];
    const float* bh2  = (const float*)d_in[6];
    const float* g2   = (const float*)d_in[7];
    const float* be2  = (const float*)d_in[8];
    const float* Wh3  = (const float*)d_in[9];
    const float* bh3  = (const float*)d_in[10];
    const float* ensw = (const float*)d_in[11];

    char* ws = (char*)d_ws;
    u16*   xbf  = (u16*)  (ws + OFF_XBF);
    float* part = (float*)(ws + OFF_PART);
    float* h2f  = (float*)(ws + OFF_H2);
    float* wsm  = (float*)(ws + OFF_WSM);
    u16*   W1bf = (u16*)  (ws + OFF_W1BF);
    float* b1f  = (float*)(ws + OFF_B1F);
    u16*   W2bf = (u16*)  (ws + OFF_W2BF);
    float* b2f  = (float*)(ws + OFF_B2F);

    // zero padded W2 region (rows 100..111 stay zero)
    hipMemsetAsync(W2bf, 0, (size_t)NSUB * NCLS_PAD * 32 * sizeof(u16), stream);

    kA<<<256, 512, 0, stream>>>(x, xbf, part);
    kB<<<1, 512, 0, stream>>>(part, Wh1, bh1, g1, be1, Wh2, bh2, g2, be2, ensw, h2f, wsm);
    kC<<<(NSUB * TOTAL) / 8, 256, 0, stream>>>(Wh3, bh3, h2f, W1bf, b1f, W2bf, b2f);
    kD<<<NROWS / 64, 128, 0, stream>>>(xbf, W1bf, b1f, W2bf, b2f, wsm, (float*)d_out);
}
